// CoreFoldingV40_17068200034780
// MI455X (gfx1250) — matrix operand order repaired and checked by reading, not compiled, not run on hardware
//
#include <hip/hip_runtime.h>
#include <math.h>

// ---------------- constants ----------------
#define NN_   768
#define DS_   384
#define DP_   128
#define DE_   64
#define DH_   512
#define HDS_  12
#define DPT_  4
#define NE_   24576
#define MIN_  832      // 2*DS + DE
#define CH_   32       // DS / HEADS
#define FD_   64       // padded per-head feature dim for logits (32 qk + 12 pts + 20 pad)
#define NN2_  (NN_ * NN_)

typedef __attribute__((ext_vector_type(16))) _Float16 v16h;
typedef __attribute__((ext_vector_type(8)))  _Float16 v8h;
typedef __attribute__((ext_vector_type(8)))  float    v8f;
typedef __attribute__((ext_vector_type(4)))  float    v4f;

// ---------------- WMMA fragment helpers (ISA §7.12.2 layouts) ----------------
// A (16x32 f16, M=lane&15): element e -> K = (e<8 ? 8*half+e : 16+8*half+(e-8))
__device__ __forceinline__ v16h load_A_frag(const _Float16* __restrict__ Arow, int k_base, int half) {
  const _Float16* p = Arow + k_base + 8 * half;
  v8h lo = *(const v8h*)(p);        // K = k_base + 8*half + [0..7]
  v8h hi = *(const v8h*)(p + 16);   // K = k_base + 16 + 8*half + [0..7]
  return __builtin_shufflevector(lo, hi, 0,1,2,3,4,5,6,7,8,9,10,11,12,13,14,15);
}

// A fragment sourced from f32 data with on-the-fly convert (used for `pair`)
__device__ __forceinline__ v16h load_A_frag_f32(const float* __restrict__ Arow, int k_base, int half) {
  const float* p = Arow + k_base + 8 * half;
  v4f a0 = *(const v4f*)(p);
  v4f a1 = *(const v4f*)(p + 4);
  v4f a2 = *(const v4f*)(p + 16);
  v4f a3 = *(const v4f*)(p + 20);
  v16h r;
#pragma unroll
  for (int i = 0; i < 4; ++i) {
    r[i]      = (_Float16)a0[i];
    r[4 + i]  = (_Float16)a1[i];
    r[8 + i]  = (_Float16)a2[i];
    r[12 + i] = (_Float16)a3[i];
  }
  return r;
}

// B (32x16 f16, N=lane&15): element e -> K = (e<8 ? 8*half+e : 16+8*half+(e-8))  (weights stored transposed [N][K])
__device__ __forceinline__ v16h load_B_frag(const _Float16* __restrict__ WtRow, int k_base, int half) {
  const _Float16* p = WtRow + k_base + 8 * half;
  v8h lo = *(const v8h*)(p);
  v8h hi = *(const v8h*)(p + 16);
  return __builtin_shufflevector(lo, hi, 0,1,2,3,4,5,6,7,8,9,10,11,12,13,14,15);
}

__device__ __forceinline__ v8f wmma32(v16h a, v16h b, v8f c) {
  return __builtin_amdgcn_wmma_f32_16x16x32_f16(false, a, false, b, (short)0, c, false, false);
}

// One wave computes a 16x16 C tile: C[m_base:+16, n_base:+16] = A[m,:K] . Wt[n,:K]
__device__ __forceinline__ v8f gemm_tile(const _Float16* __restrict__ A, int lda,
                                         const _Float16* __restrict__ Wt, int ldw,
                                         int m_base, int n_base, int K, int lane) {
  int half = lane >> 4;
  const _Float16* Arow = A + (size_t)(m_base + (lane & 15)) * lda;
  const _Float16* Wrow = Wt + (size_t)(n_base + (lane & 15)) * ldw;
  v8f acc = {};
  for (int k = 0; k < K; k += 32)
    acc = wmma32(load_A_frag(Arow, k, half), load_B_frag(Wrow, k, half), acc);
  return acc;
}

// One wave computes a 32x32 C tile (2x2 sub-tiles): 4 WMMAs per 8 b128 loads
// (double the WMMA density of the 16x16 path; used for the big edge GEMMs)
__device__ __forceinline__ void gemm_tile32(const _Float16* __restrict__ A, int lda,
                                            const _Float16* __restrict__ Wt, int ldw,
                                            int m_base, int n_base, int K, int lane,
                                            v8f acc[2][2]) {
  int half = lane >> 4;
  const _Float16* Arow0 = A + (size_t)(m_base + (lane & 15)) * lda;
  const _Float16* Arow1 = Arow0 + (size_t)16 * lda;
  const _Float16* Wrow0 = Wt + (size_t)(n_base + (lane & 15)) * ldw;
  const _Float16* Wrow1 = Wrow0 + (size_t)16 * ldw;
  for (int k = 0; k < K; k += 32) {
    if (k + 32 < K) {  // prefetch next K-step of A (global_prefetch_b8)
      __builtin_prefetch(Arow0 + k + 32, 0, 1);
      __builtin_prefetch(Arow1 + k + 32, 0, 1);
    }
    v16h a0 = load_A_frag(Arow0, k, half);
    v16h a1 = load_A_frag(Arow1, k, half);
    v16h b0 = load_B_frag(Wrow0, k, half);
    v16h b1 = load_B_frag(Wrow1, k, half);
    acc[0][0] = wmma32(a0, b0, acc[0][0]);
    acc[0][1] = wmma32(a0, b1, acc[0][1]);
    acc[1][0] = wmma32(a1, b0, acc[1][0]);
    acc[1][1] = wmma32(a1, b1, acc[1][1]);
  }
}

__device__ __forceinline__ float siluf(float v) { return v / (1.f + __expf(-v)); }

// ---------------- small utility kernels ----------------
__global__ void k_copy_f32(float* __restrict__ dst, const float* __restrict__ src, int n) {
  int t = blockIdx.x * blockDim.x + threadIdx.x;
  if (t < n) dst[t] = src[t];
}

// dst[N][K] f16  <-  src[K][Nsrc] f32  (transpose + convert, zero-pad to Ndst cols)
__global__ void k_transpose_f16(_Float16* __restrict__ dst, const float* __restrict__ src,
                                int K, int Nsrc, int Ndst) {
  int t = blockIdx.x * blockDim.x + threadIdx.x;
  if (t >= K * Ndst) return;
  int n = t / K, k = t % K;
  dst[t] = (n < Nsrc) ? (_Float16)src[(size_t)k * Nsrc + n] : (_Float16)0.f;
}

// ---------------- EGNN ----------------
__global__ void k_edge_features(const float* __restrict__ h, const float* __restrict__ x,
                                const int* __restrict__ ei,
                                const float* __restrict__ We1, const float* __restrict__ be1,
                                const float* __restrict__ We2, const float* __restrict__ be2,
                                _Float16* __restrict__ m_in) {
  int e = blockIdx.x * blockDim.x + threadIdx.x;
  if (e >= NE_) return;
  int s = ei[e], d = ei[NE_ + e];
  float dx = x[s*3+0]-x[d*3+0], dy = x[s*3+1]-x[d*3+1], dz = x[s*3+2]-x[d*3+2];
  float dist = sqrtf(dx*dx + dy*dy + dz*dz);
  float hb[DE_];
#pragma unroll 8
  for (int j = 0; j < DE_; ++j) hb[j] = siluf(dist * We1[j] + be1[j]);
  _Float16* row = m_in + (size_t)e * MIN_;
  const float* hs = h + (size_t)s * DS_;
  const float* hd = h + (size_t)d * DS_;
  for (int c = 0; c < DS_; ++c) row[c]       = (_Float16)hs[c];
  for (int c = 0; c < DS_; ++c) row[DS_ + c] = (_Float16)hd[c];
  for (int j = 0; j < DE_; ++j) {
    float a = be2[j];
#pragma unroll 8
    for (int i = 0; i < DE_; ++i) a += hb[i] * We2[i * DE_ + j];
    row[2 * DS_ + j] = (_Float16)a;
  }
}

// ---------------- WMMA GEMM kernels (8 waves / 256-thread block) ----------------
// 32x32-tile f16-out (silu option) — used for the big edge-hidden GEMMs
__global__ void k_gemm32_f16out(const _Float16* __restrict__ A, int lda,
                                const _Float16* __restrict__ Wt, int ldw,
                                const float* __restrict__ bias,
                                _Float16* __restrict__ out, int ldo,
                                int M, int Ncols, int K, int act) {
  int wave = blockIdx.x * 8 + (threadIdx.x >> 5);
  int lane = threadIdx.x & 31;
  int tilesN = Ncols >> 5, tilesM = M >> 5;
  if (wave >= tilesM * tilesN) return;
  int m_base = (wave / tilesN) << 5, n_base = (wave % tilesN) << 5;
  v8f acc[2][2] = {};
  gemm_tile32(A, lda, Wt, ldw, m_base, n_base, K, lane, acc);
  int half = lane >> 4;
#pragma unroll
  for (int ni = 0; ni < 2; ++ni) {
    int n = n_base + ni * 16 + (lane & 15);
    float bv = bias ? bias[n] : 0.f;
#pragma unroll
    for (int mi = 0; mi < 2; ++mi)
#pragma unroll
      for (int j = 0; j < 8; ++j) {
        float v = acc[mi][ni][j] + bv;
        if (act) v = siluf(v);
        out[(size_t)(m_base + mi * 16 + j + 8 * half) * ldo + n] = (_Float16)v;
      }
  }
}

// 32x32-tile scatter: h_new[dst[e]] += (A@Wt + bias)[e]
__global__ void k_gemm32_scatter(const _Float16* __restrict__ A, int lda,
                                 const _Float16* __restrict__ Wt, int ldw,
                                 const float* __restrict__ bias,
                                 const int* __restrict__ dstIdx,
                                 float* __restrict__ out,
                                 int M, int Ncols, int K) {
  int wave = blockIdx.x * 8 + (threadIdx.x >> 5);
  int lane = threadIdx.x & 31;
  int tilesN = Ncols >> 5, tilesM = M >> 5;
  if (wave >= tilesM * tilesN) return;
  int m_base = (wave / tilesN) << 5, n_base = (wave % tilesN) << 5;
  v8f acc[2][2] = {};
  gemm_tile32(A, lda, Wt, ldw, m_base, n_base, K, lane, acc);
  int half = lane >> 4;
#pragma unroll
  for (int ni = 0; ni < 2; ++ni) {
    int n = n_base + ni * 16 + (lane & 15);
    float bv = bias[n];
#pragma unroll
    for (int mi = 0; mi < 2; ++mi)
#pragma unroll
      for (int j = 0; j < 8; ++j) {
        int m = m_base + mi * 16 + j + 8 * half;
        atomicAdd(&out[(size_t)dstIdx[m] * Ncols + n], acc[mi][ni][j] + bv);
      }
  }
}

// 16x16-tile f32-out (small projections)
__global__ void k_gemm_f32out(const _Float16* __restrict__ A, int lda,
                              const _Float16* __restrict__ Wt, int ldw,
                              const float* __restrict__ bias,
                              float* __restrict__ out, int ldo,
                              int M, int Ncols, int K) {
  int wave = blockIdx.x * 8 + (threadIdx.x >> 5);
  int lane = threadIdx.x & 31;
  int tilesN = Ncols >> 4, tilesM = M >> 4;
  if (wave >= tilesM * tilesN) return;
  int m_base = (wave / tilesN) << 4, n_base = (wave % tilesN) << 4;
  v8f acc = gemm_tile(A, lda, Wt, ldw, m_base, n_base, K, lane);
  int n = n_base + (lane & 15), half = lane >> 4;
  float bv = bias ? bias[n] : 0.f;
#pragma unroll
  for (int j = 0; j < 8; ++j)
    out[(size_t)(m_base + j + 8 * half) * ldo + n] = acc[j] + bv;
}

// coord_w = hidden_c . Wc2 ; x_new[dst] += w * (x[src]-x[dst])
__global__ void k_coordw(const _Float16* __restrict__ hidc, const float* __restrict__ Wc2,
                         const float* __restrict__ x, const int* __restrict__ ei,
                         float* __restrict__ x_new) {
  int e = blockIdx.x * blockDim.x + threadIdx.x;
  if (e >= NE_) return;
  const _Float16* row = hidc + (size_t)e * DH_;
  float w = 0.f;
#pragma unroll 8
  for (int j = 0; j < DH_; ++j) w += (float)row[j] * Wc2[j];
  int s = ei[e], d = ei[NE_ + e];
#pragma unroll
  for (int i = 0; i < 3; ++i)
    atomicAdd(&x_new[d * 3 + i], w * (x[s * 3 + i] - x[d * 3 + i]));
}

__global__ void k_single_update(const float* __restrict__ single, const float* __restrict__ h_new,
                                float* __restrict__ ss, _Float16* __restrict__ sh) {
  int t = blockIdx.x * blockDim.x + threadIdx.x;
  if (t >= NN_ * DS_) return;
  float v = single[t] + h_new[t];
  ss[t] = v;
  sh[t] = (_Float16)v;
}

// ---------------- IPA ----------------
// pair_bias: pair[NN2,128] (f32, converted inline) @ WpbT[16,128] -> pb[h][i][j]
__global__ void k_pair_bias(const float* __restrict__ pair, const _Float16* __restrict__ WtPb,
                            const float* __restrict__ bpb, float* __restrict__ pb) {
  int wave = blockIdx.x * 8 + (threadIdx.x >> 5);
  int lane = threadIdx.x & 31;
  const int tilesM = NN2_ >> 4;
  if (wave >= tilesM) return;
  int m_base = wave << 4, half = lane >> 4;
  const float*    Arow = pair + (size_t)(m_base + (lane & 15)) * DP_;
  const _Float16* Wrow = WtPb + (size_t)(lane & 15) * DP_;
  v8f acc = {};
#pragma unroll
  for (int k = 0; k < DP_; k += 32)
    acc = wmma32(load_A_frag_f32(Arow, k, half), load_B_frag(Wrow, k, half), acc);
  int hcol = lane & 15;
  if (hcol < HDS_) {
    float bv = bpb[hcol];
#pragma unroll
    for (int j = 0; j < 8; ++j)
      pb[(size_t)hcol * NN2_ + m_base + j + 8 * half] = acc[j] + bv;
  }
}

// pack per-(n,h): Fq/Fk = [q(32) | rotated+translated points(12) | 0(20)] f16; q2/k2; vT[h][c][n]
__global__ void k_pack(const float* __restrict__ qf, const float* __restrict__ kf,
                       const float* __restrict__ vf,
                       const float* __restrict__ qpf, const float* __restrict__ kpf,
                       const float* __restrict__ rot, const float* __restrict__ x_new,
                       _Float16* __restrict__ Fq, _Float16* __restrict__ Fk,
                       _Float16* __restrict__ vT, float* __restrict__ q2, float* __restrict__ k2) {
  int t = blockIdx.x * blockDim.x + threadIdx.x;
  if (t >= NN_ * HDS_) return;
  int n = t / HDS_, hh = t % HDS_;
  const float* R  = rot + (size_t)n * 9;
  const float* tr = x_new + (size_t)n * 3;

  {
    const float* q = qf + (size_t)n * DS_ + hh * CH_;
    _Float16* f = Fq + ((size_t)n * HDS_ + hh) * FD_;
    for (int c = 0; c < CH_; ++c) f[c] = (_Float16)q[c];
    const float* qp = qpf + (size_t)n * (HDS_ * 3 * DPT_) + hh * (3 * DPT_);
    float ssq = 0.f;
#pragma unroll
    for (int p = 0; p < DPT_; ++p)
#pragma unroll
      for (int j = 0; j < 3; ++j) {
        float g = tr[j] + qp[0*DPT_+p]*R[0*3+j] + qp[1*DPT_+p]*R[1*3+j] + qp[2*DPT_+p]*R[2*3+j];
        f[CH_ + p * 3 + j] = (_Float16)g;
        ssq += g * g;
      }
    for (int c = CH_ + 12; c < FD_; ++c) f[c] = (_Float16)0.f;
    q2[(size_t)n * HDS_ + hh] = ssq;
  }
  {
    const float* kk = kf + (size_t)n * DS_ + hh * CH_;
    _Float16* f = Fk + ((size_t)n * HDS_ + hh) * FD_;
    for (int c = 0; c < CH_; ++c) f[c] = (_Float16)kk[c];
    const float* kp = kpf + (size_t)n * (HDS_ * 3 * DPT_) + hh * (3 * DPT_);
    float ssq = 0.f;
#pragma unroll
    for (int p = 0; p < DPT_; ++p)
#pragma unroll
      for (int j = 0; j < 3; ++j) {
        float g = tr[j] + kp[0*DPT_+p]*R[0*3+j] + kp[1*DPT_+p]*R[1*3+j] + kp[2*DPT_+p]*R[2*3+j];
        f[CH_ + p * 3 + j] = (_Float16)g;
        ssq += g * g;
      }
    for (int c = CH_ + 12; c < FD_; ++c) f[c] = (_Float16)0.f;
    k2[(size_t)n * HDS_ + hh] = ssq;
  }
  for (int c = 0; c < CH_; ++c)
    vT[((size_t)hh * CH_ + c) * NN_ + n] = (_Float16)vf[(size_t)n * DS_ + hh * CH_ + c];
}

// logits[h][n][m] = scale*(q.k + qpts.kpts) + pb - 0.5*scale*(q2+k2)  — K=64 (2 WMMAs/tile)
__global__ void k_logits(const _Float16* __restrict__ Fq, const _Float16* __restrict__ Fk,
                         const float* __restrict__ q2, const float* __restrict__ k2,
                         const float* __restrict__ pb, float* __restrict__ lg) {
  int hh = blockIdx.y;
  int wave = blockIdx.x * 8 + (threadIdx.x >> 5);
  int lane = threadIdx.x & 31;
  const int tilesN = NN_ >> 4;
  if (wave >= tilesN * tilesN) return;
  int n_base = (wave / tilesN) << 4, m_base = (wave % tilesN) << 4;
  v8f acc = gemm_tile(Fq + hh * FD_, HDS_ * FD_, Fk + hh * FD_, HDS_ * FD_,
                      n_base, m_base, FD_, lane);
  const float scale = 0.17677669529663687f;  // 1/sqrt(32)
  int m = m_base + (lane & 15), half = lane >> 4;
  float k2v = k2[(size_t)m * HDS_ + hh];
#pragma unroll
  for (int j = 0; j < 8; ++j) {
    int n = n_base + j + 8 * half;
    size_t idx = (size_t)hh * NN2_ + (size_t)n * NN_ + m;
    lg[idx] = scale * acc[j] + pb[idx] - 0.5f * scale * (q2[(size_t)n * HDS_ + hh] + k2v);
  }
}

__global__ void k_softmax(const float* __restrict__ lg, _Float16* __restrict__ attn) {
  size_t row = blockIdx.x;
  const float* L = lg + row * NN_;
  _Float16* Ao = attn + row * NN_;
  int tid = threadIdx.x;
  __shared__ float red[256];
  float m = -1e30f;
  for (int j = tid; j < NN_; j += 256) m = fmaxf(m, L[j]);
  red[tid] = m; __syncthreads();
  for (int st = 128; st > 0; st >>= 1) { if (tid < st) red[tid] = fmaxf(red[tid], red[tid + st]); __syncthreads(); }
  float mx = red[0]; __syncthreads();
  float s = 0.f;
  for (int j = tid; j < NN_; j += 256) { float e = __expf(L[j] - mx); Ao[j] = (_Float16)e; s += e; }
  red[tid] = s; __syncthreads();
  for (int st = 128; st > 0; st >>= 1) { if (tid < st) red[tid] += red[tid + st]; __syncthreads(); }
  float inv = 1.f / red[0]; __syncthreads();
  for (int j = tid; j < NN_; j += 256) Ao[j] = (_Float16)((float)Ao[j] * inv);
}

// weighted[:, h*32+c] = attn[h] @ vT[h]   (K = 768), 32x32 tiles
__global__ void k_attn_v(const _Float16* __restrict__ attn, const _Float16* __restrict__ vT,
                         _Float16* __restrict__ wout) {
  int hh = blockIdx.y;
  int wave = blockIdx.x * 8 + (threadIdx.x >> 5);
  int lane = threadIdx.x & 31;
  const int tilesM = NN_ >> 5;   // 24 tiles of 32 rows; N = 32 (one tile)
  if (wave >= tilesM) return;
  int m_base = wave << 5;
  v8f acc[2][2] = {};
  gemm_tile32(attn + (size_t)hh * NN2_, NN_, vT + (size_t)hh * CH_ * NN_, NN_,
              m_base, 0, NN_, lane, acc);
  int half = lane >> 4;
#pragma unroll
  for (int ni = 0; ni < 2; ++ni) {
    int n = ni * 16 + (lane & 15);
#pragma unroll
    for (int mi = 0; mi < 2; ++mi)
#pragma unroll
      for (int j = 0; j < 8; ++j)
        wout[(size_t)(m_base + mi * 16 + j + 8 * half) * DS_ + hh * CH_ + n] = (_Float16)acc[mi][ni][j];
  }
}

__global__ void k_layernorm(const float* __restrict__ single_s, const float* __restrict__ proj,
                            const float* __restrict__ g, const float* __restrict__ b,
                            float* __restrict__ out) {
  int n = blockIdx.x, tid = threadIdx.x;
  __shared__ float row[DS_];
  __shared__ float red[256];
  for (int c = tid; c < DS_; c += 256) row[c] = single_s[(size_t)n * DS_ + c] + proj[(size_t)n * DS_ + c];
  __syncthreads();
  float s = 0.f;
  for (int c = tid; c < DS_; c += 256) s += row[c];
  red[tid] = s; __syncthreads();
  for (int st = 128; st > 0; st >>= 1) { if (tid < st) red[tid] += red[tid + st]; __syncthreads(); }
  float mean = red[0] / (float)DS_; __syncthreads();
  float v = 0.f;
  for (int c = tid; c < DS_; c += 256) { float d = row[c] - mean; v += d * d; }
  red[tid] = v; __syncthreads();
  for (int st = 128; st > 0; st >>= 1) { if (tid < st) red[tid] += red[tid + st]; __syncthreads(); }
  float inv = rsqrtf(red[0] / (float)DS_ + 1e-5f); __syncthreads();
  for (int c = tid; c < DS_; c += 256)
    out[(size_t)n * DS_ + c] = (row[c] - mean) * inv * g[c] + b[c];
}

// ---------------- host-side orchestration ----------------
static inline size_t align256(size_t x) { return (x + 255) & ~(size_t)255; }
static inline int gemm_blocks16(int M, int N) { return ((M >> 4) * (N >> 4) + 7) / 8; }
static inline int gemm_blocks32(int M, int N) { return ((M >> 5) * (N >> 5) + 7) / 8; }

extern "C" void kernel_launch(void* const* d_in, const int* in_sizes, int n_in,
                              void* d_out, int out_size, void* d_ws, size_t ws_size,
                              hipStream_t stream) {
  (void)in_sizes; (void)n_in; (void)out_size; (void)ws_size;

  const float* h      = (const float*)d_in[0];
  const float* x      = (const float*)d_in[1];
  const int*   ei     = (const int*)d_in[2];
  const float* single = (const float*)d_in[3];
  const float* pair   = (const float*)d_in[4];
  const float* rot    = (const float*)d_in[5];
  const float* We1 = (const float*)d_in[6],  *be1 = (const float*)d_in[7];
  const float* We2 = (const float*)d_in[8],  *be2 = (const float*)d_in[9];
  const float* Wn1 = (const float*)d_in[10], *bn1 = (const float*)d_in[11];
  const float* Wn2 = (const float*)d_in[12], *bn2 = (const float*)d_in[13];
  const float* Wc1 = (const float*)d_in[14], *bc1 = (const float*)d_in[15];
  const float* Wc2 = (const float*)d_in[16];
  const float* Wq  = (const float*)d_in[17], *bq  = (const float*)d_in[18];
  const float* Wk  = (const float*)d_in[19], *bk  = (const float*)d_in[20];
  const float* Wv  = (const float*)d_in[21], *bv  = (const float*)d_in[22];
  const float* Wpb = (const float*)d_in[23], *bpb = (const float*)d_in[24];
  const float* Wqp = (const float*)d_in[25], *bqp = (const float*)d_in[26];
  const float* Wkp = (const float*)d_in[27], *bkp = (const float*)d_in[28];
  const float* Wo  = (const float*)d_in[29], *bo  = (const float*)d_in[30];
  const float* lng = (const float*)d_in[31], *lnb = (const float*)d_in[32];

  char* w = (char*)d_ws;
  size_t off = 0;
  auto take = [&](size_t bytes) -> void* { void* p = w + off; off = align256(off + bytes); return p; };

  _Float16* wt_n1 = (_Float16*)take((size_t)DH_ * MIN_ * 2);
  _Float16* wt_n2 = (_Float16*)take((size_t)DS_ * DH_ * 2);
  _Float16* wt_c1 = (_Float16*)take((size_t)DH_ * MIN_ * 2);
  _Float16* wt_q  = (_Float16*)take((size_t)DS_ * DS_ * 2);
  _Float16* wt_k  = (_Float16*)take((size_t)DS_ * DS_ * 2);
  _Float16* wt_v  = (_Float16*)take((size_t)DS_ * DS_ * 2);
  _Float16* wt_qp = (_Float16*)take((size_t)144 * DS_ * 2);
  _Float16* wt_kp = (_Float16*)take((size_t)144 * DS_ * 2);
  _Float16* wt_pb = (_Float16*)take((size_t)16 * DP_ * 2);
  _Float16* wt_o  = (_Float16*)take((size_t)DS_ * DS_ * 2);
  _Float16* m_in  = (_Float16*)take((size_t)NE_ * MIN_ * 2);
  _Float16* hid_n = (_Float16*)take((size_t)NE_ * DH_ * 2);
  _Float16* hid_c = (_Float16*)take((size_t)NE_ * DH_ * 2);
  float* h_new    = (float*)take((size_t)NN_ * DS_ * 4);
  float* x_new    = (float*)take((size_t)NN_ * 3 * 4);
  float* single_s = (float*)take((size_t)NN_ * DS_ * 4);
  _Float16* single_h = (_Float16*)take((size_t)NN_ * DS_ * 2);
  float* qf  = (float*)take((size_t)NN_ * DS_ * 4);
  float* kf  = (float*)take((size_t)NN_ * DS_ * 4);
  float* vf  = (float*)take((size_t)NN_ * DS_ * 4);
  float* qpf = (float*)take((size_t)NN_ * 144 * 4);
  float* kpf = (float*)take((size_t)NN_ * 144 * 4);
  _Float16* Fq = (_Float16*)take((size_t)NN_ * HDS_ * FD_ * 2);
  _Float16* Fk = (_Float16*)take((size_t)NN_ * HDS_ * FD_ * 2);
  _Float16* vT = (_Float16*)take((size_t)HDS_ * CH_ * NN_ * 2);
  float* q2 = (float*)take((size_t)NN_ * HDS_ * 4);
  float* k2 = (float*)take((size_t)NN_ * HDS_ * 4);
  float* pb = (float*)take((size_t)HDS_ * NN2_ * 4);
  float* lg = (float*)take((size_t)HDS_ * NN2_ * 4);
  _Float16* attn = (_Float16*)take((size_t)HDS_ * NN2_ * 2);
  _Float16* wgt  = (_Float16*)take((size_t)NN_ * DS_ * 2);
  float* proj    = (float*)take((size_t)NN_ * DS_ * 4);

  // ---- weight transpose/convert ----
  auto LT = [&](_Float16* dst, const float* src, int K, int Nsrc, int Ndst) {
    int total = K * Ndst;
    k_transpose_f16<<<(total + 255) / 256, 256, 0, stream>>>(dst, src, K, Nsrc, Ndst);
  };
  LT(wt_n1, Wn1, MIN_, DH_, DH_);
  LT(wt_n2, Wn2, DH_, DS_, DS_);
  LT(wt_c1, Wc1, MIN_, DH_, DH_);
  LT(wt_q,  Wq,  DS_, DS_, DS_);
  LT(wt_k,  Wk,  DS_, DS_, DS_);
  LT(wt_v,  Wv,  DS_, DS_, DS_);
  LT(wt_qp, Wqp, DS_, 144, 144);
  LT(wt_kp, Wkp, DS_, 144, 144);
  LT(wt_pb, Wpb, DP_, HDS_, 16);
  LT(wt_o,  Wo,  DS_, DS_, DS_);

  // ---- EGNN ----
  k_copy_f32<<<(NN_ * DS_ + 255) / 256, 256, 0, stream>>>(h_new, h, NN_ * DS_);
  k_copy_f32<<<(NN_ * 3 + 255) / 256, 256, 0, stream>>>(x_new, x, NN_ * 3);
  k_edge_features<<<(NE_ + 255) / 256, 256, 0, stream>>>(h, x, ei, We1, be1, We2, be2, m_in);
  k_gemm32_f16out<<<gemm_blocks32(NE_, DH_), 256, 0, stream>>>(m_in, MIN_, wt_n1, MIN_, bn1,
                                                               hid_n, DH_, NE_, DH_, MIN_, 1);
  k_gemm32_scatter<<<gemm_blocks32(NE_, DS_), 256, 0, stream>>>(hid_n, DH_, wt_n2, DH_, bn2,
                                                                ei + NE_, h_new, NE_, DS_, DH_);
  k_gemm32_f16out<<<gemm_blocks32(NE_, DH_), 256, 0, stream>>>(m_in, MIN_, wt_c1, MIN_, bc1,
                                                               hid_c, DH_, NE_, DH_, MIN_, 1);
  k_coordw<<<(NE_ + 255) / 256, 256, 0, stream>>>(hid_c, Wc2, x, ei, x_new);
  k_single_update<<<(NN_ * DS_ + 255) / 256, 256, 0, stream>>>(single, h_new, single_s, single_h);

  // ---- IPA projections ----
  k_gemm_f32out<<<gemm_blocks16(NN_, DS_), 256, 0, stream>>>(single_h, DS_, wt_q, DS_, bq, qf, DS_, NN_, DS_, DS_);
  k_gemm_f32out<<<gemm_blocks16(NN_, DS_), 256, 0, stream>>>(single_h, DS_, wt_k, DS_, bk, kf, DS_, NN_, DS_, DS_);
  k_gemm_f32out<<<gemm_blocks16(NN_, DS_), 256, 0, stream>>>(single_h, DS_, wt_v, DS_, bv, vf, DS_, NN_, DS_, DS_);
  k_gemm_f32out<<<gemm_blocks16(NN_, 144), 256, 0, stream>>>(single_h, DS_, wt_qp, DS_, bqp, qpf, 144, NN_, 144, DS_);
  k_gemm_f32out<<<gemm_blocks16(NN_, 144), 256, 0, stream>>>(single_h, DS_, wt_kp, DS_, bkp, kpf, 144, NN_, 144, DS_);
  k_pack<<<(NN_ * HDS_ + 255) / 256, 256, 0, stream>>>(qf, kf, vf, qpf, kpf, rot, x_new,
                                                       Fq, Fk, vT, q2, k2);
  k_pair_bias<<<(NN2_ / 16 + 7) / 8, 256, 0, stream>>>(pair, wt_pb, bpb, pb);

  // ---- attention ----
  k_logits<<<dim3((48 * 48 + 7) / 8, HDS_), 256, 0, stream>>>(Fq, Fk, q2, k2, pb, lg);
  k_softmax<<<HDS_ * NN_, 256, 0, stream>>>(lg, attn);
  k_attn_v<<<dim3((24 + 7) / 8, HDS_), 256, 0, stream>>>(attn, vT, wgt);

  // ---- output projection + residual + LayerNorm ----
  k_gemm_f32out<<<gemm_blocks16(NN_, DS_), 256, 0, stream>>>(wgt, DS_, wt_o, DS_, bo, proj, DS_, NN_, DS_, DS_);
  k_layernorm<<<NN_, 256, 0, stream>>>(single_s, proj, lng, lnb, (float*)d_out);
}
